// ASIS_50302656970939
// MI455X (gfx1250) — compile-verified, matrix operand order checked by reading, not run on hardware
//
// ASIS (PointNet++ sem/ins segmentation) forward for MI455X (gfx1250).
//
// All linear layers run as WMMA f16->f32 GEMMs (v_wmma_f32_16x16x32_f16),
// one wave per 16x16 C tile, K padded to 32, Cout padded to 16 via
// zero-filled f16 staging buffers (EXEC must be all-1s for WMMA).
// The shared 16xKpad weight tile is staged into LDS per block with the
// CDNA5 async DMA path (global_load_async_to_lds_b128 + s_wait_asynccnt)
// when the toolchain declares the builtins, else a plain LDS copy.
// Irregular stages (FPS / ball-query / 3-NN interp / kNN top-30) are
// VALU kernels; kNN uses __builtin_prefetch -> global_prefetch_b8.
//
// Input flattening assumption (compile-only, cannot verify): JAX pytree
// leaf order = alphabetical dict keys at every level, lists in order,
// per-layer dict {'w','b','g','be'} -> b,be,g,w. 137 leaves, pc last.

#include <hip/hip_runtime.h>

typedef __attribute__((ext_vector_type(16))) _Float16 v16h;
typedef __attribute__((ext_vector_type(8)))  float    v8f;
typedef __attribute__((ext_vector_type(4)))  int      v4i;

union F16x16 { v16h v; unsigned u[8]; };
union F32x8  { v8f  v; float   f[8]; };

#define BB   4
#define N0   4096
#define NS   32
#define KNN  30
#define THRE 0.5f
#define MAX_KPAD 768   // largest padded Cin (fp4 layer 1)

static __host__ __device__ inline int pad16i(int x){ return (x + 15) & ~15; }
static __host__ __device__ inline int pad32i(int x){ return (x + 31) & ~31; }

// ---- CDNA5 async global->LDS copy (ASYNCcnt) with safe fallback ----------
__device__ __forceinline__ void async_copy16(const _Float16* g, _Float16* l)
{
#if __has_builtin(__builtin_amdgcn_global_load_async_to_lds_b128)
    // builtin signature (from hipcc diagnostic): (v4i AS(1)*, v4i AS(3)*, imm, imm)
    typedef __attribute__((address_space(1))) v4i GV4;
    typedef __attribute__((address_space(3))) v4i LV4;
    __builtin_amdgcn_global_load_async_to_lds_b128(
        (GV4*)(unsigned long long)(size_t)g,
        (LV4*)(unsigned)(size_t)l,     // flat shared addr[31:0] == LDS offset
        0, 0);
#else
    ((unsigned long long*)l)[0] = ((const unsigned long long*)g)[0];
    ((unsigned long long*)l)[1] = ((const unsigned long long*)g)[1];
#endif
}

__device__ __forceinline__ void wait_async()
{
#if __has_builtin(__builtin_amdgcn_s_wait_asynccnt)
    __builtin_amdgcn_s_wait_asynccnt(0);
#elif __has_builtin(__builtin_amdgcn_global_load_async_to_lds_b128)
    asm volatile("s_wait_asynccnt 0" ::: "memory");
#endif
}

// ---------------------------------------------------------------- WMMA GEMM
// Y[M x Cout] = A[M x Kpad](f16) * W[CoutPad x Kpad](f16, row = out-ch) ^T + b
__global__ void __launch_bounds__(128)
k_wmma_gemm_bias(const _Float16* __restrict__ A, const _Float16* __restrict__ W,
                 const float* __restrict__ bias, float* __restrict__ Y,
                 int M, int Kpad, int Cout)
{
    __shared__ _Float16 sB[16 * MAX_KPAD];   // 24 KB: one weight N-tile

    // -- cooperative async stage of the (contiguous) 16 x Kpad weight tile
    {
        const _Float16* src = W + (size_t)blockIdx.y * 16 * Kpad;
        const int nchunk = (16 * Kpad) >> 3;          // 16-byte chunks
        for (int c = threadIdx.x; c < nchunk; c += 128)
            async_copy16(src + c * 8, &sB[c * 8]);
        wait_async();
    }
    __syncthreads();

    const int lane  = threadIdx.x & 31;
    const int wave  = threadIdx.x >> 5;
    const int mtile = blockIdx.x * 4 + wave;
    if (mtile * 16 >= M) return;                 // uniform per wave
    const int half = lane >> 4;                  // K-slab select (ISA 7.12.2)
    const int r    = lane & 15;                  // A row / B col within tile

    const unsigned* Arow = (const unsigned*)(A + (size_t)(mtile * 16 + r) * Kpad);
    const unsigned* Brow = (const unsigned*)(sB) + ((r * Kpad) >> 1);

    F32x8 acc;
#pragma unroll
    for (int i = 0; i < 8; ++i) acc.f[i] = 0.0f;

    for (int k0 = 0; k0 < Kpad; k0 += 32) {
        F16x16 a, b;
#pragma unroll
        for (int j = 0; j < 8; ++j) {
            // VGPR j holds K pair kj,kj+1 ; lanes>=16 get the +8 K slab
            const int kj = ((j < 4) ? (j << 1) : (16 + ((j - 4) << 1))) + (half << 3);
            a.u[j] = Arow[(k0 + kj) >> 1];
            b.u[j] = Brow[(k0 + kj) >> 1];
        }
        acc.v = __builtin_amdgcn_wmma_f32_16x16x32_f16(
                    false, a.v, false, b.v, (short)0, acc.v, false, false);
    }
    // D layout: lane<16 -> rows 0..7, lane>=16 -> rows 8..15; col = lane&15
    const int n = blockIdx.y * 16 + r;
    if (n < Cout) {
        const float bi   = bias ? bias[n] : 0.0f;
        const int  mbase = mtile * 16 + (half << 3);
#pragma unroll
        for (int v = 0; v < 8; ++v)
            Y[(size_t)(mbase + v) * Cout + n] = acc.f[v] + bi;
    }
}

// -------------------------------------------------------- staging / epilogue
__global__ void k_f32_to_f16_pad(const float* __restrict__ x, _Float16* __restrict__ xh,
                                 int M, int C, int Kpad)
{
    int t = blockIdx.x * blockDim.x + threadIdx.x;
    if (t >= M * Kpad) return;
    int m = t / Kpad, c = t - m * Kpad;
    xh[t] = (c < C) ? (_Float16)x[(size_t)m * C + c] : (_Float16)0.0f;
}

__global__ void k_w_to_f16_pad(const float* __restrict__ w, _Float16* __restrict__ wh,
                               int Cout, int Cin, int CoutPad, int Kpad)
{
    int t = blockIdx.x * blockDim.x + threadIdx.x;
    if (t >= CoutPad * Kpad) return;
    int o = t / Kpad, c = t - o * Kpad;
    wh[t] = (o < Cout && c < Cin) ? (_Float16)w[(size_t)o * Cin + c] : (_Float16)0.0f;
}

__global__ void k_bn_stats(const float* __restrict__ y, int M, int C,
                           float* __restrict__ mean, float* __restrict__ rstd)
{
    int c = blockIdx.x;
    if (c >= C) return;
    float s = 0.f, s2 = 0.f;
    for (int m = threadIdx.x; m < M; m += 256) {
        float v = y[(size_t)m * C + c];
        s += v; s2 += v * v;
    }
    __shared__ float ss[256], sq[256];
    ss[threadIdx.x] = s; sq[threadIdx.x] = s2;
    __syncthreads();
    for (int st = 128; st > 0; st >>= 1) {
        if (threadIdx.x < st) { ss[threadIdx.x] += ss[threadIdx.x + st];
                                sq[threadIdx.x] += sq[threadIdx.x + st]; }
        __syncthreads();
    }
    if (threadIdx.x == 0) {
        float mu  = ss[0] / (float)M;
        float var = sq[0] / (float)M - mu * mu;
        mean[c] = mu;
        rstd[c] = rsqrtf(var + 1e-5f);
    }
}

__global__ void k_bn_apply(float* __restrict__ y, const float* __restrict__ mean,
                           const float* __restrict__ rstd, const float* __restrict__ g,
                           const float* __restrict__ be, int total, int C, int relu)
{
    int t = blockIdx.x * blockDim.x + threadIdx.x;
    if (t >= total) return;
    int c = t % C;
    float v = (y[t] - mean[c]) * rstd[c] * g[c] + be[c];
    y[t] = relu ? fmaxf(v, 0.0f) : v;
}

// ---------------------------------------------------------------- geometry
__global__ void k_transpose_pc(const float* __restrict__ pc, float* __restrict__ xyz)
{   // (B,3,N) -> (B,N,3)
    int t = blockIdx.x * blockDim.x + threadIdx.x;
    if (t >= BB * N0 * 3) return;
    int c = t % 3, n = (t / 3) % N0, b = t / (3 * N0);
    xyz[t] = pc[((size_t)b * 3 + c) * N0 + n];
}

__global__ void k_transpose_out(const float* __restrict__ x, float* __restrict__ o, int C)
{   // (B,N,C) -> (B,C,N)
    int t = blockIdx.x * blockDim.x + threadIdx.x;
    if (t >= BB * N0 * C) return;
    int n = t % N0, c = (t / N0) % C, b = t / (N0 * C);
    o[t] = x[((size_t)b * N0 + n) * C + c];
}

__global__ void k_fps(const float* __restrict__ xyz, int N, int npoint, int* __restrict__ idx)
{
    const int b = blockIdx.x;
    xyz += (size_t)b * N * 3;
    idx += (size_t)b * npoint;
    __shared__ float sdist[4096];
    __shared__ float sv[256];
    __shared__ int   si[256];
    __shared__ int   sfar;
    const int t = threadIdx.x;
    for (int i = t; i < N; i += 256) sdist[i] = 1e10f;
    if (t == 0) sfar = 0;
    __syncthreads();
    for (int it = 0; it < npoint; ++it) {
        const int far = sfar;
        if (t == 0) idx[it] = far;
        const float cx = xyz[far * 3], cy = xyz[far * 3 + 1], cz = xyz[far * 3 + 2];
        float best = -1.0f; int bi = 0;
        for (int i = t; i < N; i += 256) {
            float dx = xyz[i * 3] - cx, dy = xyz[i * 3 + 1] - cy, dz = xyz[i * 3 + 2] - cz;
            float d = fminf(sdist[i], dx * dx + dy * dy + dz * dz);
            sdist[i] = d;
            if (d > best) { best = d; bi = i; }
        }
        sv[t] = best; si[t] = bi;
        __syncthreads();
        for (int s = 128; s > 0; s >>= 1) {
            if (t < s && sv[t + s] > sv[t]) { sv[t] = sv[t + s]; si[t] = si[t + s]; }
            __syncthreads();
        }
        if (t == 0) sfar = si[0];
        __syncthreads();
    }
}

__global__ void k_gather3(const float* __restrict__ xyz, const int* __restrict__ idx,
                          float* __restrict__ nxyz, int Nin, int S)
{
    int t = blockIdx.x * blockDim.x + threadIdx.x;
    if (t >= BB * S * 3) return;
    int c = t % 3, s = (t / 3) % S, b = t / (3 * S);
    nxyz[t] = xyz[((size_t)b * Nin + idx[b * S + s]) * 3 + c];
}

__global__ void k_ball_query(const float* __restrict__ xyz, const float* __restrict__ nxyz,
                             int Nin, int S, float r2, int* __restrict__ gidx)
{
    int g = blockIdx.x * blockDim.x + threadIdx.x;
    if (g >= BB * S) return;
    int b = g / S, s = g - b * S;
    const float* q = nxyz + ((size_t)b * S + s) * 3;
    const float* p = xyz + (size_t)b * Nin * 3;
    int* out = gidx + ((size_t)b * S + s) * NS;
    const float qx = q[0], qy = q[1], qz = q[2];
    int cnt = 0, first = 0; bool has = false;
    for (int j = 0; j < Nin && cnt < NS; ++j) {
        float dx = p[j * 3] - qx, dy = p[j * 3 + 1] - qy, dz = p[j * 3 + 2] - qz;
        if (dx * dx + dy * dy + dz * dz <= r2) {
            if (!has) { first = j; has = true; }
            out[cnt++] = j;
        }
    }
    for (; cnt < NS; ++cnt) out[cnt] = first;
}

__global__ void k_sa_concat(const float* __restrict__ xyz, const float* __restrict__ nxyz,
                            const float* __restrict__ pts, const int* __restrict__ gidx,
                            float* __restrict__ feat, int Nin, int S, int Cpts, int Cf)
{
    int t = blockIdx.x * blockDim.x + threadIdx.x;
    if (t >= BB * S * NS * Cf) return;
    int c = t % Cf;
    int k = (t / Cf) % NS;
    int s = (t / (Cf * NS)) % S;
    int b =  t / (Cf * NS * S);
    int gi = gidx[((size_t)b * S + s) * NS + k];
    float v;
    if (c < 3) v = xyz[((size_t)b * Nin + gi) * 3 + c] - nxyz[((size_t)b * S + s) * 3 + c];
    else       v = pts[((size_t)b * Nin + gi) * Cpts + (c - 3)];
    feat[t] = v;
}

__global__ void k_maxpool(const float* __restrict__ y, float* __restrict__ out, int S, int C)
{
    int t = blockIdx.x * blockDim.x + threadIdx.x;
    if (t >= BB * S * C) return;
    int c = t % C, s = (t / C) % S, b = t / (C * S);
    size_t base = ((size_t)(b * S + s)) * NS;
    float m = -3.4e38f;
    for (int k = 0; k < NS; ++k) m = fmaxf(m, y[(base + k) * C + c]);
    out[t] = m;
}

__global__ void k_interp_concat(const float* __restrict__ x1, const float* __restrict__ x2,
                                const float* __restrict__ p1, const float* __restrict__ p2,
                                float* __restrict__ out, int N1, int N2, int C1, int C2)
{
    int g = blockIdx.x * blockDim.x + threadIdx.x;
    if (g >= BB * N1) return;
    int b = g / N1, n = g - b * N1;
    const float* q = x1 + ((size_t)b * N1 + n) * 3;
    const float* P = x2 + (size_t)b * N2 * 3;
    const float qx = q[0], qy = q[1], qz = q[2];
    float d0 = 3.4e38f, d1 = 3.4e38f, d2 = 3.4e38f;
    int   i0 = 0, i1 = 0, i2 = 0;
    for (int j = 0; j < N2; ++j) {
        float dx = P[j * 3] - qx, dy = P[j * 3 + 1] - qy, dz = P[j * 3 + 2] - qz;
        float d = dx * dx + dy * dy + dz * dz;
        if      (d < d0) { d2 = d1; i2 = i1; d1 = d0; i1 = i0; d0 = d; i0 = j; }
        else if (d < d1) { d2 = d1; i2 = i1; d1 = d;  i1 = j; }
        else if (d < d2) { d2 = d;  i2 = j; }
    }
    float w0 = 1.f / (d0 + 1e-8f), w1 = 1.f / (d1 + 1e-8f), w2 = 1.f / (d2 + 1e-8f);
    float wsum = w0 + w1 + w2; w0 /= wsum; w1 /= wsum; w2 /= wsum;
    float* o = out + (size_t)g * (C1 + C2);
    const float* r1 = p1 + ((size_t)b * N1 + n) * C1;
    for (int c = 0; c < C1; ++c) o[c] = r1[c];
    const float* b2 = p2 + (size_t)b * N2 * C2;
    for (int c = 0; c < C2; ++c)
        o[C1 + c] = w0 * b2[(size_t)i0 * C2 + c]
                  + w1 * b2[(size_t)i1 * C2 + c]
                  + w2 * b2[(size_t)i2 * C2 + c];
}

__global__ void k_add(float* __restrict__ d, const float* __restrict__ a, int n)
{
    int t = blockIdx.x * blockDim.x + threadIdx.x;
    if (t < n) d[t] += a[t];
}

__global__ void k_knn_pool(const float* __restrict__ emb, const float* __restrict__ fsem,
                           float* __restrict__ pooled, int N)
{
    int g = blockIdx.x * blockDim.x + threadIdx.x;
    if (g >= BB * N) return;
    int b = g / N, i = g - b * N;
    const float* eb = emb + (size_t)b * N * 5;
    const float e0 = eb[i * 5], e1 = eb[i * 5 + 1], e2 = eb[i * 5 + 2],
                e3 = eb[i * 5 + 3], e4 = eb[i * 5 + 4];
    float dbest[KNN]; int ibest[KNN];
    for (int k = 0; k < KNN; ++k) { dbest[k] = 3.4e38f; ibest[k] = i; }
    for (int j = 0; j < N; ++j) {
        if (j + 8 < N) __builtin_prefetch(eb + (size_t)(j + 8) * 5, 0, 3);
        float a0 = eb[j * 5] - e0, a1 = eb[j * 5 + 1] - e1, a2 = eb[j * 5 + 2] - e2,
              a3 = eb[j * 5 + 3] - e3, a4 = eb[j * 5 + 4] - e4;
        float d = a0 * a0 + a1 * a1 + a2 * a2 + a3 * a3 + a4 * a4;
        if (d < dbest[KNN - 1]) {
            int k = KNN - 1;
            while (k > 0 && dbest[k - 1] > d) {
                dbest[k] = dbest[k - 1]; ibest[k] = ibest[k - 1]; --k;
            }
            dbest[k] = d; ibest[k] = j;
        }
    }
    const float* fb = fsem + (size_t)b * N * 128;
    float* po = pooled + ((size_t)b * N + i) * 128;
    for (int c = 0; c < 128; ++c) {
        float m = -3.4e38f;
        for (int k = 0; k < KNN; ++k) {
            int src = (dbest[k] < THRE) ? ibest[k] : i;   // nv > -THRE  <=>  d < THRE
            m = fmaxf(m, fb[(size_t)src * 128 + c]);
        }
        po[c] = m;
    }
}

// ------------------------------------------------------------------- host
struct LinP { const float *w, *b, *g, *be; int cin, cout; };

extern "C" void kernel_launch(void* const* d_in, const int* in_sizes, int n_in,
                              void* d_out, int out_size, void* d_ws, size_t ws_size,
                              hipStream_t stream)
{
    if (!d_out || n_in < 137) return;

    auto nb = [](long long n) { return (unsigned)((n + 255) / 256); };

    // ---- parameter leaves (alphabetical pytree order; see header comment)
    auto bnL = [&](int base, int cin, int cout) {
        LinP l; l.b = (const float*)d_in[base];     l.be = (const float*)d_in[base + 1];
                l.g = (const float*)d_in[base + 2]; l.w  = (const float*)d_in[base + 3];
        l.cin = cin; l.cout = cout; return l;
    };
    auto plL = [&](int base, int cin, int cout) {
        LinP l; l.b = (const float*)d_in[base]; l.w = (const float*)d_in[base + 1];
        l.g = nullptr; l.be = nullptr; l.cin = cin; l.cout = cout; return l;
    };
    LinP ins_fc1 = bnL(0, 128, 128);
    LinP ins_fc2 = plL(4, 128, 5);
    LinP ins_fp1[3] = { bnL(6, 131, 128), bnL(10, 128, 128), bnL(14, 128, 128) };
    LinP ins_fp2[2] = { bnL(18, 320, 256), bnL(22, 256, 128) };
    LinP ins_fp3[2] = { bnL(26, 384, 256), bnL(30, 256, 256) };
    LinP ins_fp4[2] = { bnL(34, 768, 256), bnL(38, 256, 256) };
    LinP sa1[3] = { bnL(42, 6, 32),    bnL(46, 32, 32),   bnL(50, 32, 64) };
    LinP sa2[3] = { bnL(54, 67, 64),   bnL(58, 64, 64),   bnL(62, 64, 128) };
    LinP sa3[3] = { bnL(66, 131, 128), bnL(70, 128, 128), bnL(74, 128, 256) };
    LinP sa4[3] = { bnL(78, 259, 256), bnL(82, 256, 256), bnL(86, 256, 512) };
    LinP sem_fc1 = bnL(90, 128, 128);
    LinP sem_fc2 = bnL(94, 128, 128);
    LinP sem_fc3 = plL(98, 128, 13);
    LinP sem_fp1[3] = { bnL(100, 131, 128), bnL(104, 128, 128), bnL(108, 128, 128) };
    LinP sem_fp2[2] = { bnL(112, 320, 256), bnL(116, 256, 128) };
    LinP sem_fp3[2] = { bnL(120, 384, 256), bnL(124, 256, 256) };
    LinP sem_fp4[2] = { bnL(128, 768, 256), bnL(132, 256, 256) };
    const float* pc = (const float*)d_in[136];

    // ---- workspace bump allocator (~100 MB total)
    char* base0 = (char*)d_ws;
    size_t off = 0;
    auto alloc = [&](size_t bytes) -> void* {
        off = (off + 255) & ~(size_t)255;
        void* p = base0 + off; off += bytes; return p;
    };
    float* xyz0   = (float*)alloc((size_t)BB * N0 * 3 * 4);
    float* l1x    = (float*)alloc((size_t)BB * 1024 * 3 * 4);
    float* l2x    = (float*)alloc((size_t)BB * 256 * 3 * 4);
    float* l3x    = (float*)alloc((size_t)BB * 64 * 3 * 4);
    float* l4x    = (float*)alloc((size_t)BB * 16 * 3 * 4);
    float* l1     = (float*)alloc((size_t)BB * 1024 * 64 * 4);
    float* l2     = (float*)alloc((size_t)BB * 256 * 128 * 4);
    float* l3     = (float*)alloc((size_t)BB * 64 * 256 * 4);
    float* l4     = (float*)alloc((size_t)BB * 16 * 512 * 4);
    float* p3     = (float*)alloc((size_t)BB * 64 * 256 * 4);
    float* p2     = (float*)alloc((size_t)BB * 256 * 256 * 4);
    float* p1     = (float*)alloc((size_t)BB * 1024 * 128 * 4);
    float* p0s    = (float*)alloc((size_t)BB * N0 * 128 * 4);
    float* p0i    = (float*)alloc((size_t)BB * N0 * 128 * 4);
    float* fsem   = (float*)alloc((size_t)BB * N0 * 128 * 4);
    float* fcache = (float*)alloc((size_t)BB * N0 * 128 * 4);
    float* emb5   = (float*)alloc((size_t)BB * N0 * 5 * 4);
    float* smean  = (float*)alloc(512 * 4);
    float* srstd  = (float*)alloc(512 * 4);
    float* sy0    = (float*)alloc((size_t)8388608 * 4);   // 131072 x 64 max
    float* sy1    = (float*)alloc((size_t)4194304 * 4);   // 131072 x 32 max
    int*   sidx   = (int*)  alloc((size_t)BB * 1024 * 4);
    int*   sgidx  = (int*)  alloc((size_t)BB * 1024 * NS * 4);
    _Float16* xf16 = (_Float16*)alloc((size_t)4194304 * 2); // 131072 x Kpad(32)
    _Float16* wf16 = (_Float16*)alloc((size_t)262144 * 2);  // 256 x 768 max
    (void)ws_size; (void)in_sizes; (void)out_size;

    // ---- fused linear (+optional BN/ReLU) via WMMA
    auto run_linear = [&](const float* X, int M, const LinP& L, float* Y,
                          bool bn, bool relu) {
        const int Kpad = pad32i(L.cin), CoutPad = pad16i(L.cout);
        k_f32_to_f16_pad<<<nb((long long)M * Kpad), 256, 0, stream>>>(X, xf16, M, L.cin, Kpad);
        k_w_to_f16_pad<<<nb((long long)CoutPad * Kpad), 256, 0, stream>>>(
            L.w, wf16, L.cout, L.cin, CoutPad, Kpad);
        dim3 grid((M / 16 + 3) / 4, CoutPad / 16);
        k_wmma_gemm_bias<<<grid, 128, 0, stream>>>(xf16, wf16, L.b, Y, M, Kpad, L.cout);
        if (bn) {
            k_bn_stats<<<L.cout, 256, 0, stream>>>(Y, M, L.cout, smean, srstd);
            k_bn_apply<<<nb((long long)M * L.cout), 256, 0, stream>>>(
                Y, smean, srstd, L.g, L.be, M * L.cout, L.cout, relu ? 1 : 0);
        }
    };

    auto run_mlp = [&](const float* X, int M, const LinP* Ls, int nl, float* out_final) {
        const float* cur = X;
        for (int i = 0; i < nl; ++i) {
            float* dst = (i == nl - 1) ? out_final : ((cur == sy0) ? sy1 : sy0);
            run_linear(cur, M, Ls[i], dst, true, true);
            cur = dst;
        }
    };

    auto run_sa = [&](const float* xyz, const float* pts, int Nin, int Cpts,
                      int npoint, float radius, const LinP* Ls,
                      float* nxyz, float* pooled) {
        k_fps<<<dim3(BB), dim3(256), 0, stream>>>(xyz, Nin, npoint, sidx);
        k_gather3<<<nb((long long)BB * npoint * 3), 256, 0, stream>>>(xyz, sidx, nxyz, Nin, npoint);
        k_ball_query<<<nb((long long)BB * npoint), 256, 0, stream>>>(
            xyz, nxyz, Nin, npoint, radius * radius, sgidx);
        const int Cf = 3 + Cpts;
        const long long rows = (long long)BB * npoint * NS;
        k_sa_concat<<<nb(rows * Cf), 256, 0, stream>>>(xyz, nxyz, pts, sgidx, sy1, Nin, npoint, Cpts, Cf);
        run_mlp(sy1, (int)rows, Ls, 3, sy0);
        k_maxpool<<<nb((long long)BB * npoint * Ls[2].cout), 256, 0, stream>>>(
            sy0, pooled, npoint, Ls[2].cout);
    };

    auto run_fp = [&](const float* x1, int N1, const float* x2, int N2,
                      const float* q1, int C1, const float* q2, int C2,
                      const LinP* Ls, int nl, float* out) {
        k_interp_concat<<<nb((long long)BB * N1), 256, 0, stream>>>(
            x1, x2, q1, q2, sy1, N1, N2, C1, C2);
        run_mlp(sy1, BB * N1, Ls, nl, out);
    };

    // ---- forward pass
    k_transpose_pc<<<nb((long long)BB * N0 * 3), 256, 0, stream>>>(pc, xyz0);

    run_sa(xyz0, xyz0, 4096, 3,   1024, 0.1f, sa1, l1x, l1);
    run_sa(l1x,  l1,   1024, 64,  256,  0.2f, sa2, l2x, l2);
    run_sa(l2x,  l2,   256,  128, 64,   0.4f, sa3, l3x, l3);
    run_sa(l3x,  l3,   64,   256, 16,   0.8f, sa4, l4x, l4);

    auto branch = [&](const LinP* f4, const LinP* f3, const LinP* f2, const LinP* f1,
                      float* p0) {
        run_fp(l3x, 64,   l4x, 16,   l3,   256, l4, 512, f4, 2, p3);
        run_fp(l2x, 256,  l3x, 64,   l2,   128, p3, 256, f3, 2, p2);
        run_fp(l1x, 1024, l2x, 256,  l1,   64,  p2, 256, f2, 2, p1);
        run_fp(xyz0, 4096, l1x, 1024, xyz0, 3,  p1, 128, f1, 3, p0);
    };
    branch(sem_fp4, sem_fp3, sem_fp2, sem_fp1, p0s);
    branch(ins_fp4, ins_fp3, ins_fp2, ins_fp1, p0i);

    const int Mfc = BB * N0;   // 16384
    run_linear(p0s,  Mfc, sem_fc1, fsem,   true, false);
    run_linear(fsem, Mfc, sem_fc2, fcache, true, false);
    run_linear(p0i,  Mfc, ins_fc1, sy0,    true, false);
    k_add<<<nb((long long)Mfc * 128), 256, 0, stream>>>(sy0, fcache, Mfc * 128);
    run_linear(sy0, Mfc, ins_fc2, emb5, false, false);

    k_knn_pool<<<nb((long long)BB * N0), 256, 0, stream>>>(emb5, fsem, sy1, N0);
    run_linear(sy1, Mfc, sem_fc3, sy0, false, false);

    float* dout = (float*)d_out;
    k_transpose_out<<<nb((long long)BB * 13 * N0), 256, 0, stream>>>(sy0, dout, 13);
    k_transpose_out<<<nb((long long)BB * 5 * N0), 256, 0, stream>>>(
        emb5, dout + (size_t)BB * 13 * N0, 5);
}
